// HTGNNFinal_21569325761130
// MI455X (gfx1250) — compile-verified, matrix operand order checked by reading, not run on hardware
//
#include <hip/hip_runtime.h>
#include <hip/hip_bf16.h>

// ---------------------------------------------------------------------------
// HTGNN for MI455X (gfx1250, wave32).
// Dense GEMMs: v_wmma_f32_16x16x32_bf16, 4 accumulators/wave (64x16 strip),
// A tiles staged in LDS via global_load_async_to_lds_b128 (ASYNCcnt path).
// Sparse aggregation: global_atomic_add_f32. Reductions: wave32 shfl.
// ---------------------------------------------------------------------------

typedef __attribute__((ext_vector_type(16))) __bf16 v16bf;
typedef __attribute__((ext_vector_type(8)))  float  v8f;

#define TT   6
#define NN   50000
#define NP   50048   // NN padded to a multiple of 64 for GEMM M-blocking
#define IND  256
#define HIDD 256
#define RR   4
#define EE   800000

// --------------------- CDNA5 async LDS fill helpers ------------------------

__device__ __forceinline__ void async_copy_b128(unsigned lds_off, const void* gptr) {
    // GLOBAL_LOAD_ASYNC_TO_LDS_B128: memory -> LDS, tracked by ASYNCcnt.
    asm volatile("global_load_async_to_lds_b128 %0, %1, off"
                 :: "v"(lds_off), "v"(gptr)
                 : "memory");
}

__device__ __forceinline__ void wait_async_zero() {
    asm volatile("s_wait_asynccnt 0x0" ::: "memory");
}

// ------------------------- elementwise helpers -----------------------------

__global__ void fill4_kernel(float4* __restrict__ p, size_t n4) {
    size_t i = (size_t)blockIdx.x * blockDim.x + threadIdx.x;
    if (i < n4) p[i] = make_float4(0.f, 0.f, 0.f, 0.f);
}

__global__ void to_bf16_vec8(const float* __restrict__ x, __bf16* __restrict__ y,
                             size_t n8) {
    size_t i = (size_t)blockIdx.x * blockDim.x + threadIdx.x;
    if (i >= n8) return;
    const float4* p = (const float4*)(x + i * 8);
    float4 a = p[0], b = p[1];
    union { __bf16 v[8]; uint4 u; } o;
    o.v[0] = (__bf16)a.x; o.v[1] = (__bf16)a.y; o.v[2] = (__bf16)a.z; o.v[3] = (__bf16)a.w;
    o.v[4] = (__bf16)b.x; o.v[5] = (__bf16)b.y; o.v[6] = (__bf16)b.z; o.v[7] = (__bf16)b.w;
    *(uint4*)(y + i * 8) = o.u;
}

// W: [K, O] f32 row-major  ->  Wt: [O, K] bf16 row-major (B^T layout for GEMM)
__global__ void transpose_to_bf16(const float* __restrict__ W, __bf16* __restrict__ Wt,
                                  int K, int O) {
    int idx = blockIdx.x * blockDim.x + threadIdx.x;
    if (idx >= K * O) return;
    int k = idx / O, o = idx % O;
    Wt[(size_t)o * K + k] = (__bf16)W[idx];
}

// ------------------------- WMMA bf16 GEMM ----------------------------------
// C[NP,Nc] = A[NP,K](bf16) * Bt[Nc,K]^T(bf16) + bias[Nc], optional ReLU.
// Requires: K == 256, Nc % 128 == 0, grid.x == NP/64.
// Block = 256 threads = 8 waves; wave w owns a 64(M) x 16(N) strip
// (4 accumulators sharing one B fragment per K-step).
// A tile (64 x 256 bf16, 32 KB) is filled with async LDS loads.
__global__ __launch_bounds__(256)
void gemm_bf16_wmma(const __bf16* __restrict__ A, const __bf16* __restrict__ Bt,
                    const float* __restrict__ bias, float* __restrict__ C,
                    int Nc, int K, int relu) {
    __shared__ __bf16 As[64 * 256];   // 32 KB

    const int lane  = threadIdx.x & 31;
    const int wave  = threadIdx.x >> 5;
    const int m0    = blockIdx.x * 64;
    const int tileN = blockIdx.y * 8 + wave;

    // async cooperative A-tile fill: 64 rows x 256 cols bf16, 128B per thread
    {
        int row = threadIdx.x >> 2;          // 0..63
        int kk  = (threadIdx.x & 3) * 64;    // 0,64,128,192
        const __bf16* src = A + (size_t)(m0 + row) * K + kk;
        unsigned dst = (unsigned)(size_t)(As + row * 256 + kk);
#pragma unroll
        for (int i = 0; i < 8; ++i)
            async_copy_b128(dst + i * 16, src + i * 8);
        wait_async_zero();
    }
    __syncthreads();

    const int half = lane >> 4;   // lane group 0/1
    const int mn   = lane & 15;   // row within A frag / col within B,C frag
    const int ncol = tileN * 16 + mn;

    const __bf16* brow = Bt + (size_t)ncol * K;
    __builtin_prefetch(brow, 0, 1);          // global_prefetch_b8 on B stream

    v8f acc[4] = {};
    for (int k = 0; k < K; k += 32) {
        union { v16bf v; unsigned u[8]; } b, a0, a1, a2, a3;
#pragma unroll
        for (int v = 0; v < 8; ++v) {
            // 16-bit A/B fragment K packing (ISA 7.12.2):
            // VGPR v holds K = {k0, k0+1}, k0 = (v%4)*2 + half*8 + (v/4)*16
            int k0 = k + ((v & 3) << 1) + (half << 3) + ((v >> 2) << 4);
            b.u[v]  = *(const unsigned*)(brow + k0);
            a0.u[v] = *(const unsigned*)(As + (mn +  0) * 256 + k0);
            a1.u[v] = *(const unsigned*)(As + (mn + 16) * 256 + k0);
            a2.u[v] = *(const unsigned*)(As + (mn + 32) * 256 + k0);
            a3.u[v] = *(const unsigned*)(As + (mn + 48) * 256 + k0);
        }
        acc[0] = __builtin_amdgcn_wmma_f32_16x16x32_bf16(false, a0.v, false, b.v,
                                                         (short)0, acc[0], false, false);
        acc[1] = __builtin_amdgcn_wmma_f32_16x16x32_bf16(false, a1.v, false, b.v,
                                                         (short)0, acc[1], false, false);
        acc[2] = __builtin_amdgcn_wmma_f32_16x16x32_bf16(false, a2.v, false, b.v,
                                                         (short)0, acc[2], false, false);
        acc[3] = __builtin_amdgcn_wmma_f32_16x16x32_bf16(false, a3.v, false, b.v,
                                                         (short)0, acc[3], false, false);
    }

    const float bv = bias ? bias[ncol] : 0.0f;
#pragma unroll
    for (int s = 0; s < 4; ++s) {
#pragma unroll
        for (int r = 0; r < 8; ++r) {
            // C/D layout: VGPR r, lane -> row = r + 8*half, col = lane%16
            float val = acc[s][r] + bv;
            if (relu) val = fmaxf(val, 0.0f);
            C[(size_t)(m0 + s * 16 + r + half * 8) * Nc + ncol] = val;
        }
    }
}

// ------------------------- sparse scatter-aggregate ------------------------
// m[r][row_e][:] += w_e * z[r][col_e][:]   (H = 256), one wave per edge.
__global__ __launch_bounds__(256)
void scatter_edges(const float* __restrict__ z, const int* __restrict__ erow,
                   const int* __restrict__ ecol, const float* __restrict__ ew,
                   float* __restrict__ m) {
    const int r    = blockIdx.y;
    const int lane = threadIdx.x & 31;
    const int wave = threadIdx.x >> 5;
    const int e    = blockIdx.x * 8 + wave;
    if (e >= EE) return;

    const size_t eoff  = (size_t)r * EE + e;
    const size_t rbase = (size_t)r * NP * HIDD;   // padded node stride
    const int   row = erow[eoff];
    const int   col = ecol[eoff];
    const float w   = ew[eoff];

    const float4* src = (const float4*)(z + rbase + (size_t)col * HIDD);
    float*        dst = m + rbase + (size_t)row * HIDD;
#pragma unroll
    for (int i = 0; i < 2; ++i) {
        int slot = lane + i * 32;            // 64 float4 slots cover H=256
        float4 v = src[slot];
        float* d = dst + slot * 4;
        unsafeAtomicAdd(d + 0, v.x * w);
        unsafeAtomicAdd(d + 1, v.y * w);
        unsafeAtomicAdd(d + 2, v.z * w);
        unsafeAtomicAdd(d + 3, v.w * w);
    }
}

// ------------------------- relation attention + ReLU -----------------------
// s_bf[n][:] = bf16( relu( sum_r softmax_r(<m[r][n],q>) * m[r][n][:] ) )
__global__ __launch_bounds__(256)
void attention_kernel(const float* __restrict__ msg, const float* __restrict__ q,
                      __bf16* __restrict__ s_bf) {
    const int lane = threadIdx.x & 31;
    const int wave = threadIdx.x >> 5;
    const int n    = blockIdx.x * 8 + wave;
    if (n >= NN) return;

    const int hb = lane * 8;
    float qv[8];
    {
        float4 a = *(const float4*)(q + hb);
        float4 b = *(const float4*)(q + hb + 4);
        qv[0]=a.x; qv[1]=a.y; qv[2]=a.z; qv[3]=a.w;
        qv[4]=b.x; qv[5]=b.y; qv[6]=b.z; qv[7]=b.w;
    }

    const size_t stride = (size_t)NP * HIDD;
    float mv[RR][8];
    float logit[RR];
#pragma unroll
    for (int r = 0; r < RR; ++r) {
        const float4* p = (const float4*)(msg + r * stride + (size_t)n * HIDD + hb);
        float4 a = p[0], b = p[1];
        mv[r][0]=a.x; mv[r][1]=a.y; mv[r][2]=a.z; mv[r][3]=a.w;
        mv[r][4]=b.x; mv[r][5]=b.y; mv[r][6]=b.z; mv[r][7]=b.w;
        float d = 0.f;
#pragma unroll
        for (int i = 0; i < 8; ++i) d += mv[r][i] * qv[i];
#pragma unroll
        for (int off = 16; off > 0; off >>= 1) d += __shfl_xor(d, off, 32);
        logit[r] = d;
    }

    float mx = fmaxf(fmaxf(logit[0], logit[1]), fmaxf(logit[2], logit[3]));
    float ex[RR], sum = 0.f;
#pragma unroll
    for (int r = 0; r < RR; ++r) { ex[r] = __expf(logit[r] - mx); sum += ex[r]; }
    const float inv = 1.0f / sum;

    union { __bf16 v[8]; uint4 u; } o;
#pragma unroll
    for (int i = 0; i < 8; ++i) {
        float acc = 0.f;
#pragma unroll
        for (int r = 0; r < RR; ++r) acc += (ex[r] * inv) * mv[r][i];
        o.v[i] = (__bf16)fmaxf(acc, 0.0f);
    }
    *(uint4*)(s_bf + (size_t)n * HIDD + hb) = o.u;
}

// ------------------------- GRU gates ---------------------------------------
__global__ void gru_elem(const float* __restrict__ gi, const float* __restrict__ gh,
                         float* __restrict__ h, __bf16* __restrict__ h_bf) {
    size_t idx = (size_t)blockIdx.x * blockDim.x + threadIdx.x;
    if (idx >= (size_t)NN * HIDD) return;
    size_t n = idx >> 8;
    int    j = (int)(idx & 255);
    const float* gin = gi + n * 3 * HIDD;
    const float* ghn = gh + n * 3 * HIDD;
    float ir = gin[j], iz = gin[HIDD + j], in_ = gin[2 * HIDD + j];
    float hr = ghn[j], hz = ghn[HIDD + j], hn_ = ghn[2 * HIDD + j];
    float rg = 1.0f / (1.0f + __expf(-(ir + hr)));
    float zg = 1.0f / (1.0f + __expf(-(iz + hz)));
    float ng = tanhf(in_ + rg * hn_);
    float out = (1.0f - zg) * ng + zg * h[idx];
    h[idx]    = out;
    h_bf[idx] = (__bf16)out;
}

// ------------------------- classifier GEMV ---------------------------------
__global__ __launch_bounds__(256)
void classifier_out(const float* __restrict__ h1, const float* __restrict__ W2,
                    const float* __restrict__ b2, float* __restrict__ out) {
    const int lane = threadIdx.x & 31;
    const int wave = threadIdx.x >> 5;
    const int n    = blockIdx.x * 8 + wave;
    if (n >= NN) return;
    const float* hp = h1 + (size_t)n * HIDD + lane * 8;
    const float* wp = W2 + lane * 8;
    float d = 0.f;
#pragma unroll
    for (int i = 0; i < 8; ++i) d += hp[i] * wp[i];
#pragma unroll
    for (int off = 16; off > 0; off >>= 1) d += __shfl_xor(d, off, 32);
    if (lane == 0) out[n] = d + b2[0];
}

// ------------------------- host orchestration ------------------------------

extern "C" void kernel_launch(void* const* d_in, const int* in_sizes, int n_in,
                              void* d_out, int out_size, void* d_ws, size_t ws_size,
                              hipStream_t stream) {
    const float* feat  = (const float*)d_in[0];
    const int*   erow  = (const int*)d_in[1];
    const int*   ecol  = (const int*)d_in[2];
    const float* ew    = (const float*)d_in[3];
    const float* W_rel = (const float*)d_in[4];
    const float* b_rel = (const float*)d_in[5];
    const float* rquery= (const float*)d_in[6];
    const float* W_ih  = (const float*)d_in[7];
    const float* W_hh  = (const float*)d_in[8];
    const float* b_ih  = (const float*)d_in[9];
    const float* b_hh  = (const float*)d_in[10];
    const float* W1    = (const float*)d_in[11];
    const float* b1    = (const float*)d_in[12];
    const float* W2    = (const float*)d_in[13];
    const float* b2    = (const float*)d_in[14];
    float*       out   = (float*)d_out;

    // ---- workspace carve (256B aligned); node-major buffers use NP rows ----
    char* ws = (char*)d_ws;
    auto carve = [&](size_t bytes) -> void* {
        void* p = ws;
        ws += (bytes + 255) & ~(size_t)255;
        return p;
    };
    __bf16* wrelT  = (__bf16*)carve((size_t)RR * IND * HIDD * 2);
    __bf16* wih_bf = (__bf16*)carve((size_t)3 * HIDD * HIDD * 2);
    __bf16* whh_bf = (__bf16*)carve((size_t)3 * HIDD * HIDD * 2);
    __bf16* w1T    = (__bf16*)carve((size_t)HIDD * HIDD * 2);
    __bf16* x_bf   = (__bf16*)carve((size_t)NP * IND * 2);
    __bf16* s_bf   = (__bf16*)carve((size_t)NP * HIDD * 2);
    __bf16* h_bf   = (__bf16*)carve((size_t)NP * HIDD * 2);
    float*  h      = (float*) carve((size_t)NN * HIDD * 4);
    float*  z      = (float*) carve((size_t)RR * NP * HIDD * 4);
    float*  msg    = (float*) carve((size_t)RR * NP * HIDD * 4);
    float*  gi     = (float*) carve((size_t)NP * 3 * HIDD * 4);
    float*  gh     = (float*) carve((size_t)NP * 3 * HIDD * 4);
    float*  h1     = (float*) carve((size_t)NP * HIDD * 4);

    const dim3 blk(256);

    // ---- one-time weight conversion ----
    for (int r = 0; r < RR; ++r)
        transpose_to_bf16<<<(IND * HIDD + 255) / 256, blk, 0, stream>>>(
            W_rel + (size_t)r * IND * HIDD, wrelT + (size_t)r * IND * HIDD, IND, HIDD);
    to_bf16_vec8<<<((size_t)3 * HIDD * HIDD / 8 + 255) / 256, blk, 0, stream>>>(
        W_ih, wih_bf, (size_t)3 * HIDD * HIDD / 8);
    to_bf16_vec8<<<((size_t)3 * HIDD * HIDD / 8 + 255) / 256, blk, 0, stream>>>(
        W_hh, whh_bf, (size_t)3 * HIDD * HIDD / 8);
    transpose_to_bf16<<<(HIDD * HIDD + 255) / 256, blk, 0, stream>>>(W1, w1T, HIDD, HIDD);

    // ---- init h0 = 0 (f32) and h_bf = 0 over full padded extent ----
    {
        size_t n4 = (size_t)NN * HIDD / 4;
        fill4_kernel<<<(n4 + 255) / 256, blk, 0, stream>>>((float4*)h, n4);
        size_t n4b = (size_t)NP * HIDD * 2 / 16;
        fill4_kernel<<<(n4b + 255) / 256, blk, 0, stream>>>((float4*)h_bf, n4b);
        // also zero x_bf/s_bf pad region once is unnecessary: pad rows feed only
        // pad output rows, which are never read.
    }

    for (int t = 0; t < TT; ++t) {
        // x_t -> bf16 (NN rows; pad rows unused garbage)
        size_t nx8 = (size_t)NN * IND / 8;
        to_bf16_vec8<<<(nx8 + 255) / 256, blk, 0, stream>>>(
            feat + (size_t)t * NN * IND, x_bf, nx8);

        // z[r] = x @ W_rel[r] + b_rel[r]
        for (int r = 0; r < RR; ++r)
            gemm_bf16_wmma<<<dim3(NP / 64, HIDD / 128), blk, 0, stream>>>(
                x_bf, wrelT + (size_t)r * IND * HIDD, b_rel + (size_t)r * HIDD,
                z + (size_t)r * NP * HIDD, HIDD, IND, 0);

        // msg = 0
        size_t nm4 = (size_t)RR * NP * HIDD / 4;
        fill4_kernel<<<(nm4 + 255) / 256, blk, 0, stream>>>((float4*)msg, nm4);

        // sparse scatter-aggregate (relation in grid.y)
        scatter_edges<<<dim3((EE + 7) / 8, RR), blk, 0, stream>>>(
            z, erow + (size_t)t * RR * EE, ecol + (size_t)t * RR * EE,
            ew + (size_t)t * RR * EE, msg);

        // relation attention softmax + relu -> s (bf16)
        attention_kernel<<<(NN + 7) / 8, blk, 0, stream>>>(msg, rquery, s_bf);

        // GRU GEMMs
        gemm_bf16_wmma<<<dim3(NP / 64, 3 * HIDD / 128), blk, 0, stream>>>(
            s_bf, wih_bf, b_ih, gi, 3 * HIDD, HIDD, 0);
        gemm_bf16_wmma<<<dim3(NP / 64, 3 * HIDD / 128), blk, 0, stream>>>(
            h_bf, whh_bf, b_hh, gh, 3 * HIDD, HIDD, 0);

        // GRU gates -> h, h_bf
        gru_elem<<<((size_t)NN * HIDD + 255) / 256, blk, 0, stream>>>(gi, gh, h, h_bf);
    }

    // classifier: h1 = relu(h @ W1 + b1); out = h1 @ W2 + b2
    gemm_bf16_wmma<<<dim3(NP / 64, HIDD / 128), blk, 0, stream>>>(
        h_bf, w1T, b1, h1, HIDD, HIDD, 1);
    classifier_out<<<(NN + 7) / 8, blk, 0, stream>>>(h1, W2, b2, out);
}
